// GPT2SelfAttention_81698867905262
// MI455X (gfx1250) — matrix operand order repaired and checked by reading, not compiled, not run on hardware
//
#include <hip/hip_runtime.h>
#include <hip/hip_bf16.h>
#include <stdint.h>

// ---------------------------------------------------------------------------
// GPT-2 self-attention forward for MI455X (gfx1250, wave32, WMMA, async-LDS).
// d_out = [ out : 16384*256 f32 | attn : 64*2048*2048 f32 ]
// Attention is fused: scores -> softmax -> PV live in a 128KB LDS stripe
// (possible only with CDNA5's 320KB LDS); the attn tile is pushed to HBM with
// global_store_async_from_lds_b128 overlapping the PV WMMAs.
// ---------------------------------------------------------------------------

typedef __bf16 bf16_t;
typedef __attribute__((ext_vector_type(16))) __bf16 v16bf;
typedef __attribute__((ext_vector_type(8)))  float  v8f;

#define BATCH   8
#define SEQ     2048
#define DMODEL  256
#define NHEAD   8
#define HDIM    32
#define BH      (BATCH * NHEAD)      // 64
#define MTOT    (BATCH * SEQ)        // 16384

union BF16Frag { uint4 u4[2]; v16bf v; };

__device__ inline v8f wmma_bf16(v16bf a, v16bf b, v8f c) {
  // D = A(16x32 bf16) * B(32x16 bf16) + C(16x16 f32)
  return __builtin_amdgcn_wmma_f32_16x16x32_bf16(
      /*neg_a=*/false, a, /*neg_b=*/false, b,
      /*c_mod=*/(short)0, c, /*reuse_a=*/false, /*reuse_b=*/false);
}

// A fragment: row-major [16 x 32] bf16, leading dim lda (elements).
// Lane L (<16): row L, K = {0..7, 16..23}; lane L+16: row L, K = {8..15, 24..31}.
__device__ inline v16bf load_a_frag_bf16(const bf16_t* __restrict__ base, int lda) {
  const int lane = threadIdx.x & 31;
  const int row  = lane & 15;
  const int koff = (lane >> 4) << 3;            // 0 or 8
  const bf16_t* p = base + row * lda + koff;
  BF16Frag f;
  f.u4[0] = *(const uint4*)(p);                 // K = koff .. koff+7
  f.u4[1] = *(const uint4*)(p + 16);            // K = 16+koff .. 16+koff+7
  return f.v;
}

// A fragment sourced from fp32 (LDS-resident attention probs), converted bf16.
__device__ inline v16bf load_a_frag_f32(const float* base, int lda) {
  const int lane = threadIdx.x & 31;
  const int row  = lane & 15;
  const int koff = (lane >> 4) << 3;            // 0 or 8
  const float* p = base + row * lda + koff;
  float4 a0 = *(const float4*)(p);
  float4 a1 = *(const float4*)(p + 4);
  float4 a2 = *(const float4*)(p + 16);
  float4 a3 = *(const float4*)(p + 20);
  v16bf v;
  v[0]=(bf16_t)a0.x;  v[1]=(bf16_t)a0.y;  v[2]=(bf16_t)a0.z;  v[3]=(bf16_t)a0.w;
  v[4]=(bf16_t)a1.x;  v[5]=(bf16_t)a1.y;  v[6]=(bf16_t)a1.z;  v[7]=(bf16_t)a1.w;
  v[8]=(bf16_t)a2.x;  v[9]=(bf16_t)a2.y;  v[10]=(bf16_t)a2.z; v[11]=(bf16_t)a2.w;
  v[12]=(bf16_t)a3.x; v[13]=(bf16_t)a3.y; v[14]=(bf16_t)a3.z; v[15]=(bf16_t)a3.w;
  return v;
}

// B fragment: column-major storage (column n = contiguous K, stride ldb between
// columns). Lane L (<16): col L, K = {0..7, 16..23}; lane L+16: col L, K = {8..15, 24..31}.
__device__ inline v16bf load_b_frag_bf16(const bf16_t* __restrict__ base, int ldb) {
  const int lane = threadIdx.x & 31;
  const int n    = lane & 15;
  const int koff = (lane >> 4) << 3;            // 0 or 8
  const bf16_t* p = base + (size_t)n * ldb + koff;
  BF16Frag f;
  f.u4[0] = *(const uint4*)(p);
  f.u4[1] = *(const uint4*)(p + 16);
  return f.v;
}

// ---------------------------------------------------------------------------
// Kernel 0: fp32 -> bf16 conversion
// ---------------------------------------------------------------------------
__global__ __launch_bounds__(256) void cvt_f32_to_bf16(const float* __restrict__ in,
                                                       bf16_t* __restrict__ out, int n) {
  int i = blockIdx.x * 256 + threadIdx.x;
  if (i < n) out[i] = (bf16_t)in[i];
}

// ---------------------------------------------------------------------------
// Kernel 1: QKV projection. qkv = x @ w_qkv^T + b_qkv
// M=16384, N=768, K=256. Block = 8 waves: 128 M rows x 64 N cols.
// ---------------------------------------------------------------------------
__global__ __launch_bounds__(256) void qkv_gemm(const bf16_t* __restrict__ xb,
                                                const bf16_t* __restrict__ wb,
                                                const float*  __restrict__ bias,
                                                bf16_t* __restrict__ qh,
                                                bf16_t* __restrict__ kh,
                                                bf16_t* __restrict__ vT) {
  const int m0 = blockIdx.x * 128 + (threadIdx.x >> 5) * 16;
  const int n0 = blockIdx.y * 64;
  v8f acc[4] = {};
  for (int k0 = 0; k0 < DMODEL; k0 += 32) {
    v16bf a = load_a_frag_bf16(xb + (size_t)m0 * DMODEL + k0, DMODEL);
#pragma unroll
    for (int t = 0; t < 4; ++t) {
      v16bf b = load_b_frag_bf16(wb + (size_t)(n0 + t * 16) * DMODEL + k0, DMODEL);
      acc[t] = wmma_bf16(a, b, acc[t]);
    }
  }
  const int lane  = threadIdx.x & 31;
  const int col   = lane & 15;
  const int rbase = (lane >> 4) << 3;
#pragma unroll
  for (int t = 0; t < 4; ++t) {
    const int n     = n0 + t * 16 + col;
    const float bv  = bias[n];
    const int which = n >> 8;     // 0=q 1=k 2=v
    const int r     = n & 255;
    const int h     = r >> 5;
    const int hd    = r & 31;
#pragma unroll
    for (int i = 0; i < 8; ++i) {
      const int m  = m0 + rbase + i;
      const int b_ = m >> 11;     // / SEQ
      const int s  = m & (SEQ - 1);
      const int bh = b_ * NHEAD + h;
      const bf16_t o = (bf16_t)(acc[t][i] + bv);
      if (which == 0)      qh[((size_t)bh * SEQ + s) * HDIM + hd] = o;
      else if (which == 1) kh[((size_t)bh * SEQ + s) * HDIM + hd] = o;
      else                 vT[((size_t)bh * HDIM + hd) * SEQ + s] = o;
    }
  }
}

// ---------------------------------------------------------------------------
// Kernel 2 (fused attention): per WG = one (bh, 16-query stripe).
//  Phase 1: scores (WMMA) -> LDS stripe S[16][2048] fp32 (128 KB)
//  Phase 2: row softmax in LDS (shuffle reductions)
//  Phase 3: async DMA LDS -> attn (global_store_async_from_lds_b128), no wait
//  Phase 4: PV WMMA from LDS probs (overlaps the async DMA), 8-way K-split,
//           partial reduction in LDS, ctx store; drain ASYNCcnt at the end.
// ---------------------------------------------------------------------------
__global__ __launch_bounds__(256) void attention_fused(const bf16_t* __restrict__ qh,
                                                       const bf16_t* __restrict__ kh,
                                                       const bf16_t* __restrict__ vT,
                                                       float* __restrict__ attn,
                                                       bf16_t* __restrict__ ctx) {
  __shared__ float S[16][SEQ];              // 128 KB probability stripe
  __shared__ float partial[8][16 * HDIM];   // 16 KB PV partials

  const int bh    = blockIdx.y;
  const int q0    = blockIdx.x * 16;
  const int wave  = threadIdx.x >> 5;
  const int lane  = threadIdx.x & 31;
  const int col   = lane & 15;
  const int rbase = (lane >> 4) << 3;
  const float scale = 0.17677669529663688f;   // 1/sqrt(32)

  // ---- Phase 1: scores ----
  const bf16_t* qbase = qh + ((size_t)bh * SEQ + q0) * HDIM;
  const bf16_t* kbase = kh + (size_t)bh * SEQ * HDIM;
  v16bf a = load_a_frag_bf16(qbase, HDIM);
#pragma unroll 4
  for (int t = 0; t < 16; ++t) {
    const int kt = wave * 256 + t * 16;
    if (kt > q0 + 15) {                      // tile entirely above diagonal
#pragma unroll
      for (int i = 0; i < 8; ++i) S[rbase + i][kt + col] = -10000.0f;
    } else {
      v16bf b = load_b_frag_bf16(kbase + (size_t)kt * HDIM, HDIM);
      v8f c = {};
      c = wmma_bf16(a, b, c);
#pragma unroll
      for (int i = 0; i < 8; ++i) {
        const int qi = q0 + rbase + i;
        const int ki = kt + col;
        S[rbase + i][kt + col] = (ki <= qi) ? c[i] * scale : -10000.0f;
      }
    }
  }
  __syncthreads();

  // ---- Phase 2: softmax (2 rows per wave) ----
  for (int r = wave * 2; r < wave * 2 + 2; ++r) {
    float m = -3.0e38f;
    for (int i = lane; i < SEQ; i += 32) m = fmaxf(m, S[r][i]);
#pragma unroll
    for (int off = 16; off > 0; off >>= 1) m = fmaxf(m, __shfl_xor(m, off, 32));
    float s = 0.0f;
    for (int i = lane; i < SEQ; i += 32) {
      const float v = __expf(S[r][i] - m);
      S[r][i] = v;
      s += v;
    }
#pragma unroll
    for (int off = 16; off > 0; off >>= 1) s += __shfl_xor(s, off, 32);
    const float inv = 1.0f / s;
    for (int i = lane; i < SEQ; i += 32) S[r][i] *= inv;
  }
  __syncthreads();

  // ---- Phase 3: async DMA of the 128KB prob stripe to attn (no wait yet) ----
  {
    float* dstbase = attn + (size_t)bh * SEQ * SEQ + (size_t)q0 * SEQ;
    const unsigned lds0 = (unsigned)(uintptr_t)&S[0][0];  // LDS byte offset
    for (int i = threadIdx.x; i < (16 * SEQ) / 4; i += 256) {
      const float* gdst = dstbase + (size_t)i * 4;
      const unsigned lsrc = lds0 + (unsigned)i * 16u;
      asm volatile("global_store_async_from_lds_b128 %0, %1, off"
                   :: "v"(gdst), "v"(lsrc) : "memory");
    }
  }

  // ---- Phase 4: PV, overlapping the async DMA ----
  const bf16_t* vb = vT + (size_t)bh * HDIM * SEQ;
  v8f acc0 = {}, acc1 = {};
  const int kstart = wave * 256;
  const int klim   = q0 + 16;                 // keys beyond diagonal are exact 0
  const int kend   = (kstart + 256 < klim) ? (kstart + 256) : klim;
  for (int kb = kstart; kb < kend; kb += 32) {
    v16bf pa = load_a_frag_f32(&S[0][0] + kb, SEQ);                  // ds_load
    v16bf b0 = load_b_frag_bf16(vb + kb, SEQ);                       // hd 0..15
    v16bf b1 = load_b_frag_bf16(vb + (size_t)16 * SEQ + kb, SEQ);    // hd 16..31
    acc0 = wmma_bf16(pa, b0, acc0);
    acc1 = wmma_bf16(pa, b1, acc1);
  }
#pragma unroll
  for (int i = 0; i < 8; ++i) {
    partial[wave][(rbase + i) * HDIM + col]      = acc0[i];
    partial[wave][(rbase + i) * HDIM + 16 + col] = acc1[i];
  }
  __syncthreads();

  const int b_ = bh >> 3, h = bh & 7;
  for (int e = threadIdx.x; e < 16 * HDIM; e += 256) {
    float s = 0.0f;
#pragma unroll
    for (int w = 0; w < 8; ++w) s += partial[w][e];
    const int q = e >> 5, hd = e & 31;
    ctx[((size_t)b_ * SEQ + q0 + q) * DMODEL + h * HDIM + hd] = (bf16_t)s;
  }

  asm volatile("s_wait_asynccnt 0x0" ::: "memory");   // drain attn DMA
}

// ---------------------------------------------------------------------------
// Kernel 3: out = ctx @ w_out^T + b_out. M=16384, N=256, K=256, fp32 output.
// ---------------------------------------------------------------------------
__global__ __launch_bounds__(256) void outproj_gemm(const bf16_t* __restrict__ ctx,
                                                    const bf16_t* __restrict__ wb,
                                                    const float*  __restrict__ bias,
                                                    float* __restrict__ out) {
  const int m0 = blockIdx.x * 128 + (threadIdx.x >> 5) * 16;
  const int n0 = blockIdx.y * 64;
  v8f acc[4] = {};
  for (int k0 = 0; k0 < DMODEL; k0 += 32) {
    v16bf a = load_a_frag_bf16(ctx + (size_t)m0 * DMODEL + k0, DMODEL);
#pragma unroll
    for (int t = 0; t < 4; ++t) {
      v16bf b = load_b_frag_bf16(wb + (size_t)(n0 + t * 16) * DMODEL + k0, DMODEL);
      acc[t] = wmma_bf16(a, b, acc[t]);
    }
  }
  const int lane  = threadIdx.x & 31;
  const int col   = lane & 15;
  const int rbase = (lane >> 4) << 3;
#pragma unroll
  for (int t = 0; t < 4; ++t) {
    const int n = n0 + t * 16 + col;
    const float bv = bias[n];
#pragma unroll
    for (int i = 0; i < 8; ++i) {
      const int m = m0 + rbase + i;
      out[(size_t)m * DMODEL + n] = acc[t][i] + bv;
    }
  }
}

// ---------------------------------------------------------------------------
extern "C" void kernel_launch(void* const* d_in, const int* in_sizes, int n_in,
                              void* d_out, int out_size, void* d_ws, size_t ws_size,
                              hipStream_t stream) {
  const float* x     = (const float*)d_in[0];   // [8,2048,256]
  const float* w_qkv = (const float*)d_in[1];   // [768,256]
  const float* b_qkv = (const float*)d_in[2];   // [768]
  const float* w_out = (const float*)d_in[3];   // [256,256]
  const float* b_out = (const float*)d_in[4];   // [256]

  float* out  = (float*)d_out;                              // 16384*256
  float* attn = out + (size_t)MTOT * DMODEL;                // 64*2048*2048

  char* ws = (char*)d_ws;
  size_t off = 0;
  bf16_t* xb    = (bf16_t*)(ws + off); off += (size_t)MTOT * DMODEL * 2;      // 8.0 MB
  bf16_t* wqkvb = (bf16_t*)(ws + off); off += (size_t)768 * DMODEL * 2;       // 384 KB
  bf16_t* woutb = (bf16_t*)(ws + off); off += (size_t)DMODEL * DMODEL * 2;    // 128 KB
  bf16_t* qh    = (bf16_t*)(ws + off); off += (size_t)BH * SEQ * HDIM * 2;    // 8.0 MB
  bf16_t* kh    = (bf16_t*)(ws + off); off += (size_t)BH * SEQ * HDIM * 2;    // 8.0 MB
  bf16_t* vT    = (bf16_t*)(ws + off); off += (size_t)BH * HDIM * SEQ * 2;    // 8.0 MB
  bf16_t* ctx   = (bf16_t*)(ws + off); off += (size_t)MTOT * DMODEL * 2;      // 8.0 MB
  (void)ws_size; (void)in_sizes; (void)n_in; (void)out_size;                  // ~40.5 MB total

  const int nx = MTOT * DMODEL, nwq = 768 * DMODEL, nwo = DMODEL * DMODEL;
  cvt_f32_to_bf16<<<(nx + 255) / 256, 256, 0, stream>>>(x, xb, nx);
  cvt_f32_to_bf16<<<(nwq + 255) / 256, 256, 0, stream>>>(w_qkv, wqkvb, nwq);
  cvt_f32_to_bf16<<<(nwo + 255) / 256, 256, 0, stream>>>(w_out, woutb, nwo);

  qkv_gemm<<<dim3(MTOT / 128, 768 / 64), 256, 0, stream>>>(xb, wqkvb, b_qkv, qh, kh, vT);
  attention_fused<<<dim3(SEQ / 16, BH), 256, 0, stream>>>(qh, kh, vT, attn, ctx);
  outproj_gemm<<<dim3(MTOT / 128, DMODEL / 64), 256, 0, stream>>>(ctx, woutb, b_out, out);
}
